// TtBloomAttention_23158463660694
// MI455X (gfx1250) — matrix operand order repaired and checked by reading, not compiled, not run on hardware
//
#include <hip/hip_runtime.h>
#include <hip/hip_bf16.h>

#define S_LEN 2048
#define HID   4096
#define NHEAD 32
#define HDIM  128
#define H3    (3 * HID)

typedef __bf16 bf16_t;
typedef __attribute__((ext_vector_type(16))) __bf16 v16bf;
typedef __attribute__((ext_vector_type(8)))  __bf16 v8bf;
typedef __attribute__((ext_vector_type(8)))  float  v8f;

__device__ __forceinline__ bf16_t f2bf(float f) {
  unsigned int u = __builtin_bit_cast(unsigned int, f);
  u += 0x7FFFu + ((u >> 16) & 1u);           // round-to-nearest-even
  unsigned short h = (unsigned short)(u >> 16);
  return __builtin_bit_cast(bf16_t, h);
}

__device__ __forceinline__ v8f vzero8() {
  v8f z = {0.f, 0.f, 0.f, 0.f, 0.f, 0.f, 0.f, 0.f};
  return z;
}

__device__ __forceinline__ v8f wmma_bf16(v16bf a, v16bf b, v8f c) {
  // (neg_a, A, neg_b, B, c_mod, C, reuse_a, reuse_b)
  return __builtin_amdgcn_wmma_f32_16x16x32_bf16(false, a, false, b, (short)0, c,
                                                 false, false);
}

// A fragment 16x32 (bf16): lane<16 -> row=lane, K {k0+0..7, k0+16..23};
// lane>=16 -> row=lane-16, K {k0+8..15, k0+24..31}.  `row` = lane's row pointer.
__device__ __forceinline__ v16bf load_a_bf16(const bf16_t* __restrict__ row, int k0, int half) {
  v8bf lo = *(const v8bf*)(row + k0 + half * 8);
  v8bf hi = *(const v8bf*)(row + k0 + 16 + half * 8);
  v16bf r;
#pragma unroll
  for (int i = 0; i < 8; ++i) { r[i] = lo[i]; r[8 + i] = hi[i]; }
  return r;
}

// B fragment 32x16 (bf16): lane<16 -> col=lane, K {k0+0..7, k0+16..23}; lane>=16 -> K {k0+8..15, k0+24..31}.
__device__ __forceinline__ v16bf load_b_bf16(const bf16_t* __restrict__ row, int k0, int half) {
  v8bf lo = *(const v8bf*)(row + k0 + half * 8);
  v8bf hi = *(const v8bf*)(row + k0 + 16 + half * 8);
  v16bf r;
#pragma unroll
  for (int i = 0; i < 8; ++i) { r[i] = lo[i]; r[8 + i] = hi[i]; }
  return r;
}

// ---------------------------------------------------------------------------
// Kernel 0: elementwise fp32 -> bf16 (bandwidth-bound, 8 elems/thread).
// ---------------------------------------------------------------------------
__global__ __launch_bounds__(256) void cvt_kernel(const float* __restrict__ in,
                                                  bf16_t* __restrict__ out, int n8) {
  const int i = blockIdx.x * blockDim.x + threadIdx.x;
  if (i >= n8) return;
  const int idx = i * 8;
  float4 a = *(const float4*)(in + idx);
  float4 b = *(const float4*)(in + idx + 4);
  v8bf r;
  r[0] = f2bf(a.x); r[1] = f2bf(a.y); r[2] = f2bf(a.z); r[3] = f2bf(a.w);
  r[4] = f2bf(b.x); r[5] = f2bf(b.y); r[6] = f2bf(b.z); r[7] = f2bf(b.w);
  *(v8bf*)(out + idx) = r;
}

// ---------------------------------------------------------------------------
// Kernel 1: fused QKV projection.  C[s, 3H] = X @ Wqkv^T + bqkv, scattered to
// Q[h][s][d], K[h][s][d], Vt[h][d][s]  (all bf16).
// grid = (H3/256, S/32), block = 128 (4 waves, each wave does a 32x64 tile).
// ---------------------------------------------------------------------------
__global__ __launch_bounds__(128) void qkv_kernel(
    const bf16_t* __restrict__ X, const bf16_t* __restrict__ W,
    const float* __restrict__ bias,
    bf16_t* __restrict__ Qb, bf16_t* __restrict__ Kb, bf16_t* __restrict__ Vt) {
  const int lane = threadIdx.x & 31;
  const int wave = threadIdx.x >> 5;
  const int half = lane >> 4;
  const int l15  = lane & 15;
  const int m0 = blockIdx.y * 32;
  const int n0 = blockIdx.x * 256 + wave * 64;

  const bf16_t* arow0 = X + (size_t)(m0 + l15) * HID;
  const bf16_t* arow1 = arow0 + (size_t)16 * HID;

  v8f acc[2][4];
#pragma unroll
  for (int mt = 0; mt < 2; ++mt)
#pragma unroll
    for (int nt = 0; nt < 4; ++nt) acc[mt][nt] = vzero8();

  for (int k0 = 0; k0 < HID; k0 += 32) {
    __builtin_prefetch(arow0 + k0 + 512, 0, 3);   // WGP-scope prefetch
    v16bf a0 = load_a_bf16(arow0, k0, half);
    v16bf a1 = load_a_bf16(arow1, k0, half);
#pragma unroll
    for (int nt = 0; nt < 4; ++nt) {
      const bf16_t* brow = W + (size_t)(n0 + nt * 16 + l15) * HID;
      v16bf b = load_b_bf16(brow, k0, half);
      acc[0][nt] = wmma_bf16(a0, b, acc[0][nt]);
      acc[1][nt] = wmma_bf16(a1, b, acc[1][nt]);
    }
  }

#pragma unroll
  for (int nt = 0; nt < 4; ++nt) {
    const int col = n0 + nt * 16 + l15;
    const int h = col / 384;
    const int t = (col % 384) / 128;   // 0=q, 1=k, 2=v
    const int d = col % 128;
    const float bv = bias[col];
#pragma unroll
    for (int mt = 0; mt < 2; ++mt) {
#pragma unroll
      for (int r = 0; r < 8; ++r) {
        const int row = m0 + mt * 16 + r + half * 8;
        const bf16_t v = f2bf(acc[mt][nt][r] + bv);
        if (t == 0)      Qb[((size_t)h * S_LEN + row) * HDIM + d] = v;
        else if (t == 1) Kb[((size_t)h * S_LEN + row) * HDIM + d] = v;
        else             Vt[((size_t)h * HDIM + d) * S_LEN + row] = v;
      }
    }
  }
}

// ---------------------------------------------------------------------------
// Kernel 2: flash attention with alibi + causal mask.
// grid = (S/16, NHEAD), block = 32 (one wave per 16-query tile of one head).
// ---------------------------------------------------------------------------
__global__ __launch_bounds__(32) void attn_kernel(
    const bf16_t* __restrict__ Qb, const bf16_t* __restrict__ Kb,
    const bf16_t* __restrict__ Vt, const float* __restrict__ alibi,
    bf16_t* __restrict__ Ctx) {
  __shared__ bf16_t pbuf[16 * 32];

  const int lane = threadIdx.x;
  const int half = lane >> 4;
  const int l15  = lane & 15;
  const int h  = blockIdx.y;
  const int i0 = blockIdx.x * 16;
  const float inv = 0.08838834764831845f;  // 1/sqrt(128)

  const bf16_t* qrow = Qb + ((size_t)h * S_LEN + i0 + l15) * HDIM;
  v16bf qf[4];
#pragma unroll
  for (int dq = 0; dq < 4; ++dq) qf[dq] = load_a_bf16(qrow, dq * 32, half);

  float mrow[8], lrow[8];
  v8f o[8];
#pragma unroll
  for (int r = 0; r < 8; ++r) { mrow[r] = -__builtin_inff(); lrow[r] = 0.f; }
#pragma unroll
  for (int dt = 0; dt < 8; ++dt) o[dt] = vzero8();

  const int jend = i0 + 16;  // causal: keys < i0+16 suffice for this tile
  for (int j0 = 0; j0 < jend; j0 += 32) {
    // S = Q @ K^T for 32 keys (two 16-col tiles)
    v8f s0 = vzero8(), s1 = vzero8();
    const bf16_t* krow0 = Kb + ((size_t)h * S_LEN + j0 + l15) * HDIM;
    const bf16_t* krow1 = krow0 + 16 * HDIM;
#pragma unroll
    for (int dq = 0; dq < 4; ++dq) {
      v16bf b0 = load_b_bf16(krow0, dq * 32, half);
      s0 = wmma_bf16(qf[dq], b0, s0);
      v16bf b1 = load_b_bf16(krow1, dq * 32, half);
      s1 = wmma_bf16(qf[dq], b1, s1);
    }

    const float al0 = alibi[(size_t)h * S_LEN + j0 + l15];
    const float al1 = alibi[(size_t)h * S_LEN + j0 + 16 + l15];
    float x0[8], x1[8], alpha[8];
#pragma unroll
    for (int r = 0; r < 8; ++r) {
      const int row = i0 + r + half * 8;
      const int c0 = j0 + l15;
      x0[r] = (c0      <= row) ? (s0[r] * inv + al0) : -__builtin_inff();
      x1[r] = (c0 + 16 <= row) ? (s1[r] * inv + al1) : -__builtin_inff();
    }

    // online softmax: per-row max & sum (rows live in 16-lane halves)
#pragma unroll
    for (int r = 0; r < 8; ++r) {
      float v = fmaxf(x0[r], x1[r]);
      v = fmaxf(v, __shfl_xor(v, 1));
      v = fmaxf(v, __shfl_xor(v, 2));
      v = fmaxf(v, __shfl_xor(v, 4));
      v = fmaxf(v, __shfl_xor(v, 8));
      const float mnew = fmaxf(mrow[r], v);
      alpha[r] = __expf(mrow[r] - mnew);
      mrow[r] = mnew;
      x0[r] = __expf(x0[r] - mnew);
      x1[r] = __expf(x1[r] - mnew);
      float t = x0[r] + x1[r];
      t += __shfl_xor(t, 1);
      t += __shfl_xor(t, 2);
      t += __shfl_xor(t, 4);
      t += __shfl_xor(t, 8);
      lrow[r] = lrow[r] * alpha[r] + t;
    }

    // stage P (16x32) through LDS to re-layout into an A fragment
    __syncthreads();
#pragma unroll
    for (int r = 0; r < 8; ++r) {
      const int row = r + half * 8;
      pbuf[row * 32 + l15]      = f2bf(x0[r]);
      pbuf[row * 32 + 16 + l15] = f2bf(x1[r]);
    }
    __syncthreads();
    v16bf pf;
    {
      const bf16_t* prow = &pbuf[l15 * 32];
      v8bf lo = *(const v8bf*)(prow + half * 8);
      v8bf hi = *(const v8bf*)(prow + 16 + half * 8);
#pragma unroll
      for (int i = 0; i < 8; ++i) { pf[i] = lo[i]; pf[8 + i] = hi[i]; }
    }

    // rescale running output and accumulate P @ V
#pragma unroll
    for (int dt = 0; dt < 8; ++dt) {
#pragma unroll
      for (int r = 0; r < 8; ++r) o[dt][r] *= alpha[r];
    }
#pragma unroll
    for (int dt = 0; dt < 8; ++dt) {
      const bf16_t* vrow = Vt + ((size_t)h * HDIM + dt * 16 + l15) * S_LEN;
      v16bf bv = load_b_bf16(vrow, j0, half);
      o[dt] = wmma_bf16(pf, bv, o[dt]);
    }
  }

  // normalize and write ctx in [S, H] layout (bf16) for the out projection
  float rinv[8];
#pragma unroll
  for (int r = 0; r < 8; ++r) rinv[r] = 1.f / lrow[r];
#pragma unroll
  for (int dt = 0; dt < 8; ++dt) {
#pragma unroll
    for (int r = 0; r < 8; ++r) {
      const float v = o[dt][r] * rinv[r];
      const int row = i0 + r + half * 8;
      const int col = h * HDIM + dt * 16 + l15;
      Ctx[(size_t)row * HID + col] = f2bf(v);
    }
  }
}

// ---------------------------------------------------------------------------
// Kernel 3: output projection + residual.  out = ctx @ Wd^T + bd + residual.
// grid = (HID/256, S/32), block = 128 (each wave: 32x64 tile).
// ---------------------------------------------------------------------------
__global__ __launch_bounds__(128) void proj_kernel(
    const bf16_t* __restrict__ Ctx, const bf16_t* __restrict__ Wd,
    const float* __restrict__ bd, const float* __restrict__ resid,
    float* __restrict__ out) {
  const int lane = threadIdx.x & 31;
  const int wave = threadIdx.x >> 5;
  const int half = lane >> 4;
  const int l15  = lane & 15;
  const int m0 = blockIdx.y * 32;
  const int n0 = blockIdx.x * 256 + wave * 64;

  const bf16_t* arow0 = Ctx + (size_t)(m0 + l15) * HID;
  const bf16_t* arow1 = arow0 + (size_t)16 * HID;

  v8f acc[2][4];
#pragma unroll
  for (int mt = 0; mt < 2; ++mt)
#pragma unroll
    for (int nt = 0; nt < 4; ++nt) acc[mt][nt] = vzero8();

  for (int k0 = 0; k0 < HID; k0 += 32) {
    __builtin_prefetch(arow0 + k0 + 512, 0, 3);   // WGP-scope prefetch
    v16bf a0 = load_a_bf16(arow0, k0, half);
    v16bf a1 = load_a_bf16(arow1, k0, half);
#pragma unroll
    for (int nt = 0; nt < 4; ++nt) {
      const bf16_t* brow = Wd + (size_t)(n0 + nt * 16 + l15) * HID;
      v16bf b = load_b_bf16(brow, k0, half);
      acc[0][nt] = wmma_bf16(a0, b, acc[0][nt]);
      acc[1][nt] = wmma_bf16(a1, b, acc[1][nt]);
    }
  }

#pragma unroll
  for (int nt = 0; nt < 4; ++nt) {
    const int col = n0 + nt * 16 + l15;
    const float bv = bd[col];
#pragma unroll
    for (int mt = 0; mt < 2; ++mt) {
#pragma unroll
      for (int r = 0; r < 8; ++r) {
        const int row = m0 + mt * 16 + r + half * 8;
        const size_t idx = (size_t)row * HID + col;
        out[idx] = acc[mt][nt][r] + bv + resid[idx];
      }
    }
  }
}

extern "C" void kernel_launch(void* const* d_in, const int* in_sizes, int n_in,
                              void* d_out, int out_size, void* d_ws, size_t ws_size,
                              hipStream_t stream) {
  (void)in_sizes; (void)n_in; (void)out_size; (void)ws_size;
  const float* hidden   = (const float*)d_in[0];
  const float* residual = (const float*)d_in[1];
  const float* alibi    = (const float*)d_in[2];
  // d_in[3] = attention_mask (causal) — applied analytically, unused.
  const float* Wqkv = (const float*)d_in[4];
  const float* bqkv = (const float*)d_in[5];
  const float* Wd   = (const float*)d_in[6];
  const float* bd   = (const float*)d_in[7];
  float* out = (float*)d_out;

  char* ws = (char*)d_ws;
  size_t off = 0;
  bf16_t* Xb    = (bf16_t*)(ws + off); off += (size_t)S_LEN * HID * 2;       // 16 MiB
  bf16_t* Wqkvb = (bf16_t*)(ws + off); off += (size_t)H3 * HID * 2;          // 96 MiB
  bf16_t* Wdb   = (bf16_t*)(ws + off); off += (size_t)HID * HID * 2;         // 32 MiB
  bf16_t* Qb    = (bf16_t*)(ws + off); off += (size_t)NHEAD * S_LEN * HDIM * 2;
  bf16_t* Kb    = (bf16_t*)(ws + off); off += (size_t)NHEAD * S_LEN * HDIM * 2;
  bf16_t* Vt    = (bf16_t*)(ws + off); off += (size_t)NHEAD * S_LEN * HDIM * 2;
  bf16_t* Ctx   = (bf16_t*)(ws + off); off += (size_t)S_LEN * HID * 2;

  const int nX    = S_LEN * HID / 8;
  const int nWqkv = H3 * HID / 8;
  const int nWd   = HID * HID / 8;
  cvt_kernel<<<(nX    + 255) / 256, 256, 0, stream>>>(hidden, Xb, nX);
  cvt_kernel<<<(nWqkv + 255) / 256, 256, 0, stream>>>(Wqkv, Wqkvb, nWqkv);
  cvt_kernel<<<(nWd   + 255) / 256, 256, 0, stream>>>(Wd, Wdb, nWd);

  qkv_kernel<<<dim3(H3 / 256, S_LEN / 32), 128, 0, stream>>>(Xb, Wqkvb, bqkv,
                                                             Qb, Kb, Vt);
  attn_kernel<<<dim3(S_LEN / 16, NHEAD), 32, 0, stream>>>(Qb, Kb, Vt, alibi, Ctx);
  proj_kernel<<<dim3(HID / 256, S_LEN / 32), 128, 0, stream>>>(Ctx, Wdb, bd,
                                                               residual, out);
}
